// RWKVAudioBlock_24816321036516
// MI455X (gfx1250) — compile-verified, matrix operand order checked by reading, not run on hardware
//
#include <hip/hip_runtime.h>
#include <hip/hip_bf16.h>

// ---------------------------------------------------------------------------
// RWKV audio block for MI455X (gfx1250), wave32 + WMMA f32_16x16x32_f16,
// async global->LDS double-buffered GEMM staging (ASYNCcnt).
// Pipeline:
//   0. one-time fp32->f16 conversion of all weights (Wr,Wv,Wo,Wck,Wcv)
//   1. h = LN1(x)
//   2. hs = shift(h) -> f16 ; gate = sigmoid(xx@Wg)
//   3. r  = hs @ Wr^T   (WMMA, f16 out)            [Wk GEMM skipped: k unused]
//   4. v  = hs @ Wv^T   (WMMA, fp32 -> d_out[v_first])
//   5. fwd/bwd half-decay scans + gate combine -> xcomb f16
//   6. x1 = x + xcomb @ Wo^T (WMMA, residual epilogue)
//   7. h2 = LN2(x1); k2 = shift(h2) -> f16
//   8. act = relu(k2 @ Wck^T)^2 (WMMA, relu^2 epilogue, f16)
//   9. out = x1 + act @ Wcv^T  (WMMA, residual epilogue, fp32 -> d_out)
// ---------------------------------------------------------------------------

typedef __attribute__((ext_vector_type(16))) _Float16 v16h;
typedef __attribute__((ext_vector_type(8)))  _Float16 v8h;
typedef __attribute__((ext_vector_type(8)))  float    v8f;
typedef __attribute__((ext_vector_type(4)))  float    v4f;
typedef __attribute__((ext_vector_type(4)))  int      v4i;

static constexpr int kB = 8, kT = 2048, kC = 1024, kH = 16, kN = 64;
static constexpr int BM = 128, BN = 128, BK = 32;

enum GemmMode { MODE_F32 = 0, MODE_RES = 1, MODE_F16 = 2, MODE_RELU2 = 3 };

#if defined(__has_builtin)
#if __has_builtin(__builtin_amdgcn_global_load_async_to_lds_b128) && \
    __has_builtin(__builtin_amdgcn_s_wait_asynccnt)
#define USE_ASYNC_LDS 1
#endif
#endif

// --------------------- fp32 -> f16 weight conversion -----------------------
__global__ __launch_bounds__(256) void f32_to_f16_kernel(
    const float* __restrict__ src, _Float16* __restrict__ dst, long n)
{
    const long i = ((long)blockIdx.x * 256 + threadIdx.x) * 4;
    if (i >= n) return;
    v4f w = *(const v4f*)(src + i);
    dst[i + 0] = (_Float16)w[0];
    dst[i + 1] = (_Float16)w[1];
    dst[i + 2] = (_Float16)w[2];
    dst[i + 3] = (_Float16)w[3];
}

// -------------------------- LayerNorm (per token) --------------------------
__global__ __launch_bounds__(256) void ln_kernel(
    const float* __restrict__ x, const float* __restrict__ w,
    const float* __restrict__ b, float* __restrict__ out, int C)
{
    const int token = blockIdx.x;
    const int tid   = threadIdx.x;
    const float* px = x + (size_t)token * C;
    float* po       = out + (size_t)token * C;
    __shared__ float red[256];

    float s = 0.f;
    for (int c = tid; c < C; c += 256) s += px[c];
    red[tid] = s; __syncthreads();
    for (int off = 128; off > 0; off >>= 1) {
        if (tid < off) red[tid] += red[tid + off];
        __syncthreads();
    }
    const float mu = red[0] / (float)C;
    __syncthreads();

    float s2 = 0.f;
    for (int c = tid; c < C; c += 256) { float d = px[c] - mu; s2 += d * d; }
    red[tid] = s2; __syncthreads();
    for (int off = 128; off > 0; off >>= 1) {
        if (tid < off) red[tid] += red[tid + off];
        __syncthreads();
    }
    const float inv = rsqrtf(red[0] / (float)C + 1e-5f);

    for (int c = tid; c < C; c += 256)
        po[c] = (px[c] - mu) * inv * w[c] + b[c];
}

// ------------- time-shift -> f16, plus per-head gate = sig(xx@Wg) ----------
__global__ __launch_bounds__(256) void shift_gate_kernel(
    const float* __restrict__ h, const float* __restrict__ w_g,
    _Float16* __restrict__ hs16, float* __restrict__ gate,
    int T, int C, int H)
{
    const int token = blockIdx.x;        // b*T + t
    const int t     = token % T;
    const int tid   = threadIdx.x;
    const float* hp = h + (size_t)token * C;

    __shared__ float xxs[kC];
    __shared__ float red[256];

    // phase 1: hs = h[t-1] (or 0), xx = hs - h[t]
    const int c0 = tid * 4;
    #pragma unroll
    for (int q = 0; q < 4; ++q) {
        const int c   = c0 + q;
        const float hv  = hp[c];
        const float hsv = (t == 0) ? 0.f : hp[c - C];
        hs16[(size_t)token * C + c] = (_Float16)hsv;
        xxs[c] = hsv - hv;
    }
    __syncthreads();

    // phase 2: 16 heads x 16 chunks of 64
    const int head  = tid >> 4;
    const int chunk = tid & 15;
    const float* wg = w_g + (size_t)head * C + chunk * 64;
    const float* xc = xxs + chunk * 64;
    float p = 0.f;
    #pragma unroll 8
    for (int q = 0; q < 64; ++q) p += xc[q] * wg[q];
    red[tid] = p; __syncthreads();

    if (tid < H) {
        float s = 0.f;
        #pragma unroll
        for (int q = 0; q < 16; ++q) s += red[tid * 16 + q];
        gate[(size_t)token * H + tid] = 1.f / (1.f + __expf(-s));
    }
}

// ------------------- plain time-shift (fp32 -> f16) ------------------------
__global__ __launch_bounds__(256) void shift_f16_kernel(
    const float* __restrict__ h, _Float16* __restrict__ out,
    int T, int C, long total)
{
    const long i = (long)blockIdx.x * blockDim.x + threadIdx.x;
    if (i >= total) return;
    const long token = i / C;
    const int  t     = (int)(token % T);
    out[i] = (t == 0) ? (_Float16)0.f : (_Float16)h[i - C];
}

// ------- forward/backward half-decay scans + gated combine with r ----------
__global__ __launch_bounds__(256) void wkv_scan_kernel(
    const float* __restrict__ v, const _Float16* __restrict__ r16,
    const float* __restrict__ gate, float* __restrict__ fwd,
    _Float16* __restrict__ xcomb, int T, int C, int H, int N)
{
    const int id = blockIdx.x * blockDim.x + threadIdx.x;  // 0 .. B*C-1
    const int b  = id / C;
    const int c  = id % C;
    const int h  = c / N;
    const size_t base  = (size_t)b * T * C + c;
    const size_t gbase = (size_t)b * T * H + h;

    // forward: s0 = v0; s_t = 0.5*(s_{t-1} + v_t)
    float s = v[base];
    for (int t = 0; t < T; ++t) {
        const size_t idx = base + (size_t)t * C;
        s = 0.5f * (s + v[idx]);
        fwd[idx] = s;
    }
    // backward + combine: xcomb = r * (g*fwd + (1-g)*bwd)
    s = v[base + (size_t)(T - 1) * C];
    for (int t = T - 1; t >= 0; --t) {
        const size_t idx = base + (size_t)t * C;
        s = 0.5f * (s + v[idx]);
        const float g  = gate[gbase + (size_t)t * H];
        const float rr = (float)r16[idx];
        xcomb[idx] = (_Float16)(rr * (g * fwd[idx] + (1.f - g) * s));
    }
}

// ------------------------------ WMMA GEMM ----------------------------------
// out[m,n] = sum_k A[m,k] * W[n,k]   (A, W f16 row-major, inner dim K)

__device__ __forceinline__ void async_cp_b128(const _Float16* g, _Float16* l)
{
#if defined(USE_ASYNC_LDS) && defined(__AMDGCN__)
    __builtin_amdgcn_global_load_async_to_lds_b128(
        (__attribute__((address_space(1))) v4i*)g,
        (__attribute__((address_space(3))) v4i*)l, 0, 0);
#else
    *(v8h*)l = *(const v8h*)g;   // fallback: synchronous copy through VGPRs
#endif
}

__device__ __forceinline__ void wait_async_4()
{
#if defined(USE_ASYNC_LDS) && defined(__AMDGCN__)
    __builtin_amdgcn_s_wait_asynccnt(4);
#endif
}

__device__ __forceinline__ void wait_async_0()
{
#if defined(USE_ASYNC_LDS) && defined(__AMDGCN__)
    __builtin_amdgcn_s_wait_asynccnt(0);
#endif
}

__device__ __forceinline__ v16h frag_ld(const _Float16* tile, int lane)
{
    // 16-bit A/B operand layout: lane%16 = row, lane/16 selects K-half;
    // per lane: K in [hi*8, hi*8+8) and [hi*8+16, hi*8+24)  (BK=32, stride 32)
    const int l  = lane & 15;
    const int hi = (lane >> 4) << 3;
    const _Float16* row = tile + l * BK;
    union { v16h v; v8h h[2]; } u;
    u.h[0] = *(const v8h*)(row + hi);
    u.h[1] = *(const v8h*)(row + hi + 16);
    return u.v;
}

__global__ __launch_bounds__(256) void gemm_nt_kernel(
    const _Float16* __restrict__ A, const _Float16* __restrict__ W,
    float* __restrict__ outF, _Float16* __restrict__ outH,
    const float* __restrict__ res, int M, int N, int K, int mode)
{
    __shared__ _Float16 As[2][BM * BK];
    __shared__ _Float16 Bs[2][BN * BK];

    const int tid  = threadIdx.x;
    const int lane = tid & 31;
    const int wave = tid >> 5;       // 0..7
    const int wr   = wave >> 2;      // 0..1  (64-row band)
    const int wc   = wave & 3;       // 0..3  (32-col band)
    const int rowBase = blockIdx.y * BM;
    const int colBase = blockIdx.x * BN;

    v8f acc[4][2];
    const v8f zero = {};
    #pragma unroll
    for (int i = 0; i < 4; ++i)
        #pragma unroll
        for (int j = 0; j < 2; ++j) acc[i][j] = zero;

    const int ldRow  = tid >> 1;          // 0..127
    const int ldHalf = (tid & 1) * 16;    // 0 or 16 (halves)
    const _Float16* aSrc = A + (size_t)(rowBase + ldRow) * K + ldHalf;
    const _Float16* bSrc = W + (size_t)(colBase + ldRow) * K + ldHalf;
    _Float16* aDst0 = &As[0][ldRow * BK + ldHalf];
    _Float16* bDst0 = &Bs[0][ldRow * BK + ldHalf];
    _Float16* aDst1 = &As[1][ldRow * BK + ldHalf];
    _Float16* bDst1 = &Bs[1][ldRow * BK + ldHalf];

    auto issue = [&](int k0, int buf) {
        _Float16* ad = buf ? aDst1 : aDst0;
        _Float16* bd = buf ? bDst1 : bDst0;
        async_cp_b128(aSrc + k0,     ad);
        async_cp_b128(aSrc + k0 + 8, ad + 8);
        async_cp_b128(bSrc + k0,     bd);
        async_cp_b128(bSrc + k0 + 8, bd + 8);
    };

    const int nk = K / BK;
    issue(0, 0);

    for (int kk = 0; kk < nk; ++kk) {
        const int cur = kk & 1;
        if (kk + 1 < nk) {
            issue((kk + 1) * BK, cur ^ 1);   // prefetch next slab (other buffer)
            wait_async_4();                  // tile kk's 4 copies done
        } else {
            wait_async_0();
        }
        __syncthreads();                     // tile kk visible from all waves

        v16h af[4], bf[2];
        #pragma unroll
        for (int i = 0; i < 4; ++i)
            af[i] = frag_ld(&As[cur][(wr * 64 + i * 16) * BK], lane);
        #pragma unroll
        for (int j = 0; j < 2; ++j)
            bf[j] = frag_ld(&Bs[cur][(wc * 32 + j * 16) * BK], lane);

        #pragma unroll
        for (int i = 0; i < 4; ++i)
            #pragma unroll
            for (int j = 0; j < 2; ++j)
                acc[i][j] = __builtin_amdgcn_wmma_f32_16x16x32_f16(
                    false, af[i], false, bf[j], (short)0, acc[i][j],
                    false, false);

        __syncthreads();                     // all reads done before buf reuse
    }

    // epilogue: C/D layout -> lane%16 = n, VGPR r -> m = r + (lane/16)*8
    const int l16 = lane & 15;
    const int mh  = (lane >> 4) * 8;
    #pragma unroll
    for (int i = 0; i < 4; ++i) {
        #pragma unroll
        for (int j = 0; j < 2; ++j) {
            const int n = colBase + wc * 32 + j * 16 + l16;
            #pragma unroll
            for (int r = 0; r < 8; ++r) {
                const int m = rowBase + wr * 64 + i * 16 + mh + r;
                const size_t idx = (size_t)m * N + n;
                const float a = acc[i][j][r];
                if (mode == MODE_F32) {
                    outF[idx] = a;
                } else if (mode == MODE_RES) {
                    outF[idx] = res[idx] + a;
                } else if (mode == MODE_F16) {
                    outH[idx] = (_Float16)a;
                } else {  // MODE_RELU2
                    const float t = fmaxf(a, 0.f);
                    outH[idx] = (_Float16)(t * t);
                }
            }
        }
    }
}

// ---------------------------------------------------------------------------
extern "C" void kernel_launch(void* const* d_in, const int* in_sizes, int n_in,
                              void* d_out, int out_size, void* d_ws, size_t ws_size,
                              hipStream_t stream)
{
    (void)in_sizes; (void)n_in; (void)out_size; (void)ws_size;

    const float* x     = (const float*)d_in[0];
    // d_in[1] = v_first (unused for layer 0)
    const float* ln1_w = (const float*)d_in[2];
    const float* ln1_b = (const float*)d_in[3];
    const float* ln2_w = (const float*)d_in[4];
    const float* ln2_b = (const float*)d_in[5];
    const float* w_r   = (const float*)d_in[6];
    // d_in[7] = w_k (never used by _wkv -> GEMM skipped)
    const float* w_v   = (const float*)d_in[8];
    const float* w_o   = (const float*)d_in[9];
    const float* w_g   = (const float*)d_in[10];
    const float* ck_w  = (const float*)d_in[11];
    const float* cv_w  = (const float*)d_in[12];

    const size_t MT = (size_t)kB * kT;        // tokens (16384)
    const size_t MC = MT * kC;                // token-channel elements
    const size_t CC = (size_t)kC * kC;        // 1M weight elements

    float* out1 = (float*)d_out;              // x1 + ffn
    float* vout = out1 + MC;                  // v_first

    char* p = (char*)d_ws;
    float*     h_f32 = (float*)p;     p += MC * sizeof(float);     // LN out (reused)
    _Float16*  hs16  = (_Float16*)p;  p += MC * sizeof(_Float16);  // shifted (reused)
    _Float16*  r16   = (_Float16*)p;  p += MC * sizeof(_Float16);
    float*     gate  = (float*)p;     p += MT * kH * sizeof(float);
    float*     fwd   = (float*)p;     p += MC * sizeof(float);
    _Float16*  xcomb = (_Float16*)p;  p += MC * sizeof(_Float16);
    float*     x1    = (float*)p;     p += MC * sizeof(float);
    _Float16*  act   = (_Float16*)p;  p += MT * 4 * kC * sizeof(_Float16);
    _Float16*  wr16  = (_Float16*)p;  p += CC * sizeof(_Float16);
    _Float16*  wv16  = (_Float16*)p;  p += CC * sizeof(_Float16);
    _Float16*  wo16  = (_Float16*)p;  p += CC * sizeof(_Float16);
    _Float16*  ck16  = (_Float16*)p;  p += 4 * CC * sizeof(_Float16);
    _Float16*  cv16  = (_Float16*)p;

    const dim3 blk(256);
    const dim3 gemm_g_c (kC  / BN, (int)(MT / BM));   // N=1024
    const dim3 gemm_g_4c(4*kC / BN, (int)(MT / BM));  // N=4096
    const int  cvt_c  = (int)(CC / (256 * 4));        // 1M elems, 4/thread
    const int  cvt_4c = 4 * cvt_c;

    // ---- one-time weight conversion fp32 -> f16 ----
    f32_to_f16_kernel<<<cvt_c,  blk, 0, stream>>>(w_r,  wr16, (long)CC);
    f32_to_f16_kernel<<<cvt_c,  blk, 0, stream>>>(w_v,  wv16, (long)CC);
    f32_to_f16_kernel<<<cvt_c,  blk, 0, stream>>>(w_o,  wo16, (long)CC);
    f32_to_f16_kernel<<<cvt_4c, blk, 0, stream>>>(ck_w, ck16, (long)(4 * CC));
    f32_to_f16_kernel<<<cvt_4c, blk, 0, stream>>>(cv_w, cv16, (long)(4 * CC));

    // ---- TimeMix ----
    ln_kernel<<<(int)MT, blk, 0, stream>>>(x, ln1_w, ln1_b, h_f32, kC);
    shift_gate_kernel<<<(int)MT, blk, 0, stream>>>(h_f32, w_g, hs16, gate, kT, kC, kH);

    gemm_nt_kernel<<<gemm_g_c, blk, 0, stream>>>(hs16, wr16, nullptr, r16, nullptr,
                                                 (int)MT, kC, kC, MODE_F16);
    gemm_nt_kernel<<<gemm_g_c, blk, 0, stream>>>(hs16, wv16, vout, nullptr, nullptr,
                                                 (int)MT, kC, kC, MODE_F32);

    wkv_scan_kernel<<<(kB * kC) / 256, blk, 0, stream>>>(vout, r16, gate, fwd, xcomb,
                                                         kT, kC, kH, kN);

    gemm_nt_kernel<<<gemm_g_c, blk, 0, stream>>>(xcomb, wo16, x1, nullptr, x,
                                                 (int)MT, kC, kC, MODE_RES);

    // ---- ChannelMix ----
    ln_kernel<<<(int)MT, blk, 0, stream>>>(x1, ln2_w, ln2_b, h_f32, kC);
    shift_f16_kernel<<<(int)((MC + 255) / 256), blk, 0, stream>>>(h_f32, hs16, kT, kC,
                                                                  (long)MC);

    gemm_nt_kernel<<<gemm_g_4c, blk, 0, stream>>>(hs16, ck16, nullptr, act, nullptr,
                                                  (int)MT, 4 * kC, kC, MODE_RELU2);
    gemm_nt_kernel<<<gemm_g_c, blk, 0, stream>>>(act, cv16, out1, nullptr, x1,
                                                 (int)MT, kC, 4 * kC, MODE_RES);
}
